// BiBloSAN_25778393710754
// MI455X (gfx1250) — matrix operand order repaired and checked by reading, not compiled, not run on hardware
//
#include <hip/hip_runtime.h>
#include <hip/hip_bf16.h>

// ---------------------------------------------------------------------------
// BiBloSAN on MI455X (gfx1250). wave32. All GEMMs run f16 WMMA (16x16x32,
// f32 accum) from LDS-staged tiles with coalesced b128 global loads and
// register double-buffering. Attention uses a one-pass bounded softmax
// (C*tanh => logits <= C). All f16 operands are pre-converted once; the
// whole working set (~28 MB) is resident in the 192 MB L2.
// ---------------------------------------------------------------------------

typedef __attribute__((ext_vector_type(16))) _Float16 v16h;
typedef __attribute__((ext_vector_type(8)))  float    v8f;

#define D_DIM 256
#define B_DIM 4
#define N_DIM 16
#define R_DIM 64

// ---------------------------------------------------------------------------
// WMMA GEMM: C = act(A[MxK]@W[KxN] + bias + addin), A/W in f16, out f32/f16.
// Block: ROWS/16 waves (ROWS*2 threads); tile ROWS x 64, k-step 32.
// LDS: A row-major (stride 40 halves), B transposed n-major (stride 40) so
//   A fragment = two contiguous 16B runs  (K = 8h..8h+7, 16+8h..16+8h+7)
//   B fragment = two contiguous 16B runs  (K = 8h..8h+7, 16+8h..16+8h+7) per lane
// -> all fragment reads are ds_load_b128 with conflict-free 20-dword stride.
// ---------------------------------------------------------------------------
template<int ROWS, int ACT, bool HAS_BIAS, bool HAS_ADD, bool OUT_F32, bool OUT_F16>
__global__ __launch_bounds__(ROWS * 2)
void wmma_gemm_k(const _Float16* __restrict__ A, const _Float16* __restrict__ W,
                 const float* __restrict__ bias, const float* __restrict__ addin,
                 float* __restrict__ Cf32, _Float16* __restrict__ Cf16,
                 int M, int K, int Nn)
{
    constexpr int T   = ROWS * 2;      // threads (ROWS/16 waves)
    constexpr int LDA = 40;            // padded row stride (halves)
    constexpr int LDB = 40;
    constexpr int APT = 2;             // A uint4 chunks / thread (ROWS*4 / T)
    constexpr int BPT = 256 / T;       // B uint4 chunks / thread

    __shared__ _Float16 lsA[ROWS * LDA];
    __shared__ _Float16 lsB[64 * LDB];

    const int tid  = threadIdx.x;
    const int lane = tid & 31;
    const int wave = tid >> 5;
    const int half = (lane >> 4) & 1;
    const int l16  = lane & 15;
    const int row0 = blockIdx.x * ROWS;
    const int col0 = blockIdx.y * 64;

    uint4 ra[APT], rb[BPT];
    v8f acc[4] = {};

    auto load_tiles = [&](int k0) {
#pragma unroll
        for (int u = 0; u < APT; ++u) {          // A: ROWS x 32 halves
            const int c = tid + u * T;
            const int r = c >> 2, p = c & 3;
            ra[u] = *(const uint4*)(A + (size_t)(row0 + r) * K + k0 + p * 8);
        }
#pragma unroll
        for (int u = 0; u < BPT; ++u) {          // B: 32 x 64 halves
            const int c = tid + u * T;
            const int k = c >> 3, p = c & 7;
            rb[u] = *(const uint4*)(W + (size_t)(k0 + k) * Nn + col0 + p * 8);
        }
    };
    auto store_tiles = [&]() {
#pragma unroll
        for (int u = 0; u < APT; ++u) {
            const int c = tid + u * T;
            const int r = c >> 2, p = c & 3;
            *(uint4*)(&lsA[r * LDA + p * 8]) = ra[u];
        }
#pragma unroll
        for (int u = 0; u < BPT; ++u) {          // transpose-scatter B into LDS
            const int c = tid + u * T;
            const int k = c >> 3, p = c & 7;
            union { uint4 q; _Float16 h[8]; } cv; cv.q = rb[u];
#pragma unroll
            for (int e = 0; e < 8; ++e)
                lsB[(p * 8 + e) * LDB + k] = cv.h[e];
        }
    };

    const int KT = K >> 5;
    load_tiles(0);
    store_tiles();
    __syncthreads();

    for (int kt = 0; kt < KT; ++kt) {
        if (kt + 1 < KT) load_tiles((kt + 1) << 5);   // overlap next tile fetch

        union { v16h v; uint4 q[2]; } af;
        const _Float16* aB = &lsA[(wave * 16 + l16) * LDA];
        af.q[0] = *(const uint4*)(aB + 8 * half);
        af.q[1] = *(const uint4*)(aB + 16 + 8 * half);
#pragma unroll
        for (int t = 0; t < 4; ++t) {
            union { v16h v; uint4 q[2]; } bf;
            const _Float16* bB = &lsB[(t * 16 + l16) * LDB + 8 * half];
            bf.q[0] = *(const uint4*)(bB);
            bf.q[1] = *(const uint4*)(bB + 16);
            acc[t] = __builtin_amdgcn_wmma_f32_16x16x32_f16(
                /*neg_a=*/false, af.v, /*neg_b=*/false, bf.v,
                /*c_mod=*/(short)0, acc[t], /*reuse_a=*/false, /*reuse_b=*/false);
        }
        __syncthreads();
        if (kt + 1 < KT) { store_tiles(); __syncthreads(); }
    }

    // ---- fused epilogue: C/D element r -> row wave*16 + r + 8*half, col l16
#pragma unroll
    for (int t = 0; t < 4; ++t) {
        const int col = col0 + t * 16 + l16;
        const float bv = HAS_BIAS ? bias[col] : 0.0f;
#pragma unroll
        for (int r = 0; r < 8; ++r) {
            const int row = row0 + wave * 16 + r + 8 * half;
            float val = acc[t][r] + bv;
            if (HAS_ADD) val += addin[(size_t)row * Nn + col];
            if (ACT == 1) val = fmaxf(val, 0.0f);                 // relu
            if (ACT == 2) val = 1.0f / (1.0f + __expf(-val));     // sigmoid
            if (OUT_F32) Cf32[(size_t)row * Nn + col] = val;
            if (OUT_F16) Cf16[(size_t)row * Nn + col] = (_Float16)val;
        }
    }
}

// ---------------------------------------------------------------------------
// f32 -> f16 one-shot conversion (weights / x)
// ---------------------------------------------------------------------------
__global__ __launch_bounds__(256)
void cvt_f16_k(const float* __restrict__ src, _Float16* __restrict__ dst, int n)
{
    const int i = blockIdx.x * 256 + threadIdx.x;
    if (i < n) dst[i] = (_Float16)src[i];
}

// ---------------------------------------------------------------------------
// Masked per-feature attention (mSA core), streaming over j.
// f = C*tanh((Xi+Xj+b)/C) bounded by C=5 => stable one-pass exp(f-C) softmax.
// Writes f32 H and (optionally) an f16 copy for the next GEMM.
// grid = (groups, L), block = 256 (d lanes).
// ---------------------------------------------------------------------------
__global__ __launch_bounds__(D_DIM)
void msa_attn_k(const float* __restrict__ Xi, const float* __restrict__ Xj,
                const float* __restrict__ Vals, const float* __restrict__ bvec,
                float* __restrict__ Hout, _Float16* __restrict__ HoutH,
                int L, int fw)
{
    const int g = blockIdx.x;
    const int i = blockIdx.y;
    const int d = threadIdx.x;
    const size_t row = (size_t)g * L * D_DIM + (size_t)i * D_DIM + d;

    const bool valid = fw ? (i < L - 1) : (i > 0);
    if (!valid) {
        Hout[row] = 0.0f;
        if (HoutH) HoutH[row] = (_Float16)0.0f;
        return;
    }
    const size_t base = (size_t)g * L * D_DIM;
    const float xi = Xi[base + (size_t)i * D_DIM + d] + bvec[d];
    const int jlo = fw ? i + 1 : 0;
    const int jhi = fw ? L : i;          // exclusive

    float num = 0.0f, den = 0.0f;
    for (int j = jlo; j < jhi; ++j) {
        if (j + 4 < jhi) {               // gfx1250 global_prefetch_b8
            __builtin_prefetch(&Xj[base + (size_t)(j + 4) * D_DIM + d], 0, 1);
            __builtin_prefetch(&Vals[base + (size_t)(j + 4) * D_DIM + d], 0, 1);
        }
        const float f = 5.0f * tanhf((xi + Xj[base + (size_t)j * D_DIM + d]) * 0.2f);
        const float w = __expf(f - 5.0f);
        num += w * Vals[base + (size_t)j * D_DIM + d];
        den += w;
    }
    const float hv = num / den;
    Hout[row] = hv;
    if (HoutH) HoutH[row] = (_Float16)hv;
}

// ---------------------------------------------------------------------------
// source2token: a = softmax_R(S[.,r,d]); V[g,d] = sum_r a*H. f32 + f16 out.
// ---------------------------------------------------------------------------
__global__ __launch_bounds__(D_DIM)
void s2t_reduce_k(const float* __restrict__ S, const float* __restrict__ Hv,
                  float* __restrict__ Vout, _Float16* __restrict__ VoutH)
{
    const int g = blockIdx.x;
    const int d = threadIdx.x;
    const size_t base = (size_t)g * R_DIM * D_DIM;

    float m = -3.4e38f;
    for (int r = 0; r < R_DIM; ++r)
        m = fmaxf(m, S[base + (size_t)r * D_DIM + d]);

    float num = 0.0f, den = 0.0f;
    for (int r = 0; r < R_DIM; ++r) {
        const float w = __expf(S[base + (size_t)r * D_DIM + d] - m);
        num += w * Hv[base + (size_t)r * D_DIM + d];
        den += w;
    }
    const float v = num / den;
    Vout[(size_t)g * D_DIM + d]  = v;
    VoutH[(size_t)g * D_DIM + d] = (_Float16)v;
}

// e = G*o + (1-G)*v
__global__ __launch_bounds__(D_DIM)
void gate_combine_k(const float* __restrict__ G, const float* __restrict__ O,
                    const float* __restrict__ V, float* __restrict__ E)
{
    const size_t idx = (size_t)blockIdx.x * D_DIM + threadIdx.x;
    const float g = G[idx];
    E[idx] = g * O[idx] + (1.0f - g) * V[idx];
}

// cat = [xf, hf, E] in f16 (GEMM input). xf/hf = rows (b, n=0, r=t); E=e[b,0].
__global__ __launch_bounds__(D_DIM)
void build_cat_k(const float* __restrict__ INp, const float* __restrict__ H,
                 const float* __restrict__ E, _Float16* __restrict__ Cat)
{
    const int b = blockIdx.x >> 4;
    const int t = blockIdx.x & 15;
    const int d = threadIdx.x;
    const size_t xr = ((size_t)b * N_DIM * R_DIM + t) * D_DIM;
    const size_t cr = (size_t)blockIdx.x * (3 * D_DIM);
    Cat[cr + d]             = (_Float16)INp[xr + d];
    Cat[cr + D_DIM + d]     = (_Float16)H[xr + d];
    Cat[cr + 2 * D_DIM + d] = (_Float16)E[((size_t)b * N_DIM) * D_DIM + d];
}

// u = Gf*fusion + (1-Gf)*xf -> output halves (fw | bw concat on last axis)
__global__ __launch_bounds__(D_DIM)
void final_out_k(const float* __restrict__ Gf, const float* __restrict__ Fu,
                 const float* __restrict__ INp, float* __restrict__ out, int dir)
{
    const int b = blockIdx.x >> 4;
    const int t = blockIdx.x & 15;
    const int d = threadIdx.x;
    const size_t row = (size_t)blockIdx.x;
    const float g  = Gf[row * D_DIM + d];
    const float f  = Fu[row * D_DIM + d];
    const float xv = INp[((size_t)b * N_DIM * R_DIM + t) * D_DIM + d];
    out[row * (2 * D_DIM) + dir * D_DIM + d] = g * f + (1.0f - g) * xv;
}

// ---------------------------------------------------------------------------
extern "C" void kernel_launch(void* const* d_in, const int* in_sizes, int n_in,
                              void* d_out, int out_size, void* d_ws, size_t ws_size,
                              hipStream_t stream)
{
    (void)in_sizes; (void)n_in; (void)out_size; (void)ws_size;
    const float* x = (const float*)d_in[0];
    float* out = (float*)d_out;

    constexpr int BN = B_DIM * N_DIM;       // 64
    constexpr int M  = BN * R_DIM;          // 4096
    constexpr int D  = D_DIM;

    // ---- f32 scratch ----
    float* f = (float*)d_ws;
    float* IN = f; f += (size_t)M * D;
    float* XI = f; f += (size_t)M * D;
    float* XJ = f; f += (size_t)M * D;
    float* H  = f; f += (size_t)M * D;
    float* Sb = XI;                          // XI dead after msa_attn
    float* V  = f; f += (size_t)BN * D;
    float* VI = f; f += (size_t)BN * D;
    float* VJ = f; f += (size_t)BN * D;
    float* O  = f; f += (size_t)BN * D;
    float* T1 = f; f += (size_t)BN * D;
    float* G  = f; f += (size_t)BN * D;
    float* E  = f; f += (size_t)BN * D;
    float* FU = f; f += (size_t)BN * D;
    float* GF = f; f += (size_t)BN * D;
    // ---- f16 scratch (16B-aligned: all prior counts are multiples of 4) ----
    _Float16* hp = (_Float16*)f;
    _Float16* Xh   = hp; hp += (size_t)M * D;
    _Float16* INh  = hp; hp += (size_t)M * D;
    _Float16* Hh   = hp; hp += (size_t)M * D;
    _Float16* Fbh  = hp; hp += (size_t)M * D;
    _Float16* Vh   = hp; hp += (size_t)BN * D;
    _Float16* Oh   = hp; hp += (size_t)BN * D;
    _Float16* CATh = hp; hp += (size_t)64 * 3 * D;
    _Float16* Wh[9];
    for (int i = 0; i < 7; ++i) { Wh[i] = hp; hp += (size_t)D * D; }     // DxD
    for (int i = 7; i < 9; ++i) { Wh[i] = hp; hp += (size_t)3 * D * D; } // 3DxD

    const dim3 gBig(M / 128, D / 64), bBig(256);   // 8-wave blocks, 128x64 tiles
    const dim3 gSm (BN / 64, D / 64), bSm (128);   // 4-wave blocks, 64x64 tiles

    // x -> f16 once
    cvt_f16_k<<<dim3((M * D) / 256), dim3(256), 0, stream>>>(x, Xh, M * D);

    for (int dir = 0; dir < 2; ++dir) {
        const float* const* P = (const float* const*)(d_in + 1 + dir * 16);
        const float *fcW = P[0], *fcb = P[1], *mW1 = P[2], *mW2 = P[3], *mb = P[4],
                    *s2tW1 = P[5], *s2tb1 = P[6], *s2tW = P[7], *s2tb = P[8],
                    *gW1 = P[9], *gW2 = P[10], *gb = P[11],
                    *fW1 = P[12], *fb1 = P[13], *fW2 = P[14], *fb2 = P[15];
        const int fw = (dir == 0);

        // weights -> f16 (reused buffers per direction)
        const float* wsrc[9] = { fcW, mW1, mW2, s2tW1, s2tW, gW1, gW2, fW1, fW2 };
        for (int i = 0; i < 7; ++i)
            cvt_f16_k<<<dim3((D * D) / 256), dim3(256), 0, stream>>>(wsrc[i], Wh[i], D * D);
        for (int i = 7; i < 9; ++i)
            cvt_f16_k<<<dim3((3 * D * D) / 256), dim3(256), 0, stream>>>(wsrc[i], Wh[i], 3 * D * D);
        _Float16 *fcWh = Wh[0], *mW1h = Wh[1], *mW2h = Wh[2], *s2tW1h = Wh[3],
                 *s2tWh = Wh[4], *gW1h = Wh[5], *gW2h = Wh[6], *fW1h = Wh[7], *fW2h = Wh[8];

        // in = relu(x@fcW + fcb)              (f32 + f16 outputs)
        wmma_gemm_k<128,1,true ,false,true ,true ><<<gBig, bBig, 0, stream>>>(Xh,  fcWh,  fcb,  nullptr, IN, INh, M, D, D);
        // Xi = in@mW1 ; Xj = in@mW2           (f32 only)
        wmma_gemm_k<128,0,false,false,true ,false><<<gBig, bBig, 0, stream>>>(INh, mW1h, nullptr, nullptr, XI, nullptr, M, D, D);
        wmma_gemm_k<128,0,false,false,true ,false><<<gBig, bBig, 0, stream>>>(INh, mW2h, nullptr, nullptr, XJ, nullptr, M, D, D);
        // h = mSA(in) over R within each (b,n) block
        msa_attn_k<<<dim3(BN, R_DIM), D, 0, stream>>>(XI, XJ, IN, mb, H, Hh, R_DIM, fw);
        // s2t: F = relu(h@W1+b1) (f16 only); S = F@Wo+bo (f32); v = softmax_R . h
        wmma_gemm_k<128,1,true ,false,false,true ><<<gBig, bBig, 0, stream>>>(Hh,  s2tW1h, s2tb1, nullptr, nullptr, Fbh, M, D, D);
        wmma_gemm_k<128,0,true ,false,true ,false><<<gBig, bBig, 0, stream>>>(Fbh, s2tWh,  s2tb,  nullptr, Sb, nullptr, M, D, D);
        s2t_reduce_k<<<dim3(BN), D, 0, stream>>>(Sb, H, V, Vh);
        // o = mSA(v) over the N=16 blocks per batch
        wmma_gemm_k<64,0,false,false,true ,false><<<gSm, bSm, 0, stream>>>(Vh, mW1h, nullptr, nullptr, VI, nullptr, BN, D, D);
        wmma_gemm_k<64,0,false,false,true ,false><<<gSm, bSm, 0, stream>>>(Vh, mW2h, nullptr, nullptr, VJ, nullptr, BN, D, D);
        msa_attn_k<<<dim3(B_DIM, N_DIM), D, 0, stream>>>(VI, VJ, V, mb, O, Oh, N_DIM, fw);
        // G = sigmoid(o@gW1 + v@gW2 + gb); e = G*o + (1-G)*v
        wmma_gemm_k<64,0,false,false,true ,false><<<gSm, bSm, 0, stream>>>(Oh, gW1h, nullptr, nullptr, T1, nullptr, BN, D, D);
        wmma_gemm_k<64,2,true ,true ,true ,false><<<gSm, bSm, 0, stream>>>(Vh, gW2h, gb,      T1,      G,  nullptr, BN, D, D);
        gate_combine_k<<<dim3(BN), D, 0, stream>>>(G, O, V, E);
        // fusion: cat=[xf,hf,E] (64x768 f16); u = Gf*relu(cat@fW1+fb1)+(1-Gf)*xf
        build_cat_k<<<dim3(B_DIM * 16), D, 0, stream>>>(IN, H, E, CATh);
        wmma_gemm_k<64,1,true ,false,true ,false><<<gSm, bSm, 0, stream>>>(CATh, fW1h, fb1, nullptr, FU, nullptr, 64, 3 * D, D);
        wmma_gemm_k<64,2,true ,false,true ,false><<<gSm, bSm, 0, stream>>>(CATh, fW2h, fb2, nullptr, GF, nullptr, 64, 3 * D, D);
        final_out_k<<<dim3(B_DIM * 16), D, 0, stream>>>(GF, FU, IN, out, dir);
    }
}
